// GraphLookup_62818191671786
// MI455X (gfx1250) — matrix operand order repaired and checked by reading, not compiled, not run on hardware
//
#include <hip/hip_runtime.h>
#include <hip/hip_bf16.h>
#include <math.h>

typedef __attribute__((ext_vector_type(16))) _Float16 v16h;
typedef __attribute__((ext_vector_type(8)))  _Float16 v8h;
typedef __attribute__((ext_vector_type(8)))  float    v8f;

#define N_PIX   8192
#define C_DIM   512
#define K_CLS   27
#define K_PAD   32
#define H_DIM   64
#define W_DIM   64
#define RESOLUTION 0.05f

// ---- workspace layout (bytes) ----
#define OFF_MH   0                         // f16  M (N x 512)          8 MB
#define OFF_MF   (OFF_MH  + 8388608)       // f32  M (N x 512)         16 MB
#define OFF_SH   (OFF_MF  + 16777216)      // f16  S (N x 32, padded)  512 KB
#define OFF_SF   (OFF_SH  + 524288)        // f32  S (N x 32, padded)    1 MB
#define OFF_NC   (OFF_SF  + 1048576)       // f32  normed clusters (27x512)
#define OFF_MAX  (OFF_NC  + 55296)         // f32  per-pixel max inner (N)
#define OFF_STS  (OFF_MAX + 32768)         // f32  StS (729) (+pad)
#define OFF_DEG  (OFF_STS + 4096)          // f32  Deg (N)   [zeroed each call]
#define OFF_MOD  (OFF_DEG + 32768)         // f32  tr(S^T P S) scalar [zeroed]

// ------------------------------------------------------------------
// fragment loaders (row-major f16 matrix, row stride ldm elements)
// A (16x32): lane = row; elems 0..7 <- K = 8*half.., elems 8..15 <- K = 16+8*half..
// B (32x16): lane = col; elems 0..7 <- K = 8*half.., elems 8..15 <- K = 16+8*half..
// ------------------------------------------------------------------
__device__ __forceinline__ v16h load_fragA(const _Float16* M, int ldm, int rowBase,
                                           int kc, int lane) {
  const int half = lane >> 4;
  const _Float16* p = M + (size_t)(rowBase + (lane & 15)) * ldm + kc;
  v8h lo = *(const v8h*)(p + 8 * half);
  v8h hi = *(const v8h*)(p + 16 + 8 * half);
  v16h r;
#pragma unroll
  for (int i = 0; i < 8; ++i) { r[i] = lo[i]; r[i + 8] = hi[i]; }
  return r;
}

__device__ __forceinline__ v16h load_fragB(const _Float16* M, int ldm, int colBase,
                                           int kc, int lane) {
  const int half = lane >> 4;
  const _Float16* p = M + (size_t)(colBase + (lane & 15)) * ldm + kc;
  v8h lo = *(const v8h*)(p + 8 * half);
  v8h hi = *(const v8h*)(p + 16 + 8 * half);
  v16h r;
#pragma unroll
  for (int i = 0; i < 8; ++i) { r[i] = lo[i]; r[i + 8] = hi[i]; }
  return r;
}

__device__ __forceinline__ v8f wmma_f16(v16h a, v16h b, v8f c) {
  return __builtin_amdgcn_wmma_f32_16x16x32_f16(false, a, false, b,
                                                (short)0, c, false, false);
}

__device__ __forceinline__ float block_reduce256(float v, float* red) {
  const int tid = threadIdx.x;
  red[tid] = v; __syncthreads();
#pragma unroll
  for (int s = 128; s > 0; s >>= 1) {
    if (tid < s) red[tid] += red[tid + s];
    __syncthreads();
  }
  float r = red[0]; __syncthreads();
  return r;
}

// ------------------------------------------------------------------
// 1) normalize clusters: (27 x 512)
// ------------------------------------------------------------------
__global__ __launch_bounds__(256) void k_cluster_norm(const float* __restrict__ clusters,
                                                      float* __restrict__ nc) {
  __shared__ float red[256];
  const int k = blockIdx.x, tid = threadIdx.x;
  float a = clusters[k * C_DIM + tid];
  float b = clusters[k * C_DIM + tid + 256];
  float s = block_reduce256(a * a + b * b, red);
  float scale = 1.0f / fmaxf(sqrtf(s), 1e-12f);
  nc[k * C_DIM + tid]       = a * scale;
  nc[k * C_DIM + tid + 256] = b * scale;
}

// ------------------------------------------------------------------
// 2) normalize features over channels; n = b*4096 + w*64 + h (torch permute 0,3,2,1)
// ------------------------------------------------------------------
__global__ __launch_bounds__(256) void k_feat_norm(const float* __restrict__ x,
                                                   _Float16* __restrict__ Mh,
                                                   float* __restrict__ Mf) {
  __shared__ float red[256];
  const int n = blockIdx.x, tid = threadIdx.x;
  const int b = n >> 12, w = (n >> 6) & 63, h = n & 63;
  const size_t base = (size_t)b * C_DIM * H_DIM * W_DIM + h * W_DIM + w;
  float a0 = x[base + (size_t)tid * 4096];
  float a1 = x[base + (size_t)(tid + 256) * 4096];
  float s = block_reduce256(a0 * a0 + a1 * a1, red);
  float scale = 1.0f / fmaxf(sqrtf(s), 1e-12f);
  float f0 = a0 * scale, f1 = a1 * scale;
  Mf[(size_t)n * C_DIM + tid]       = f0;
  Mf[(size_t)n * C_DIM + tid + 256] = f1;
  Mh[(size_t)n * C_DIM + tid]       = (_Float16)f0;
  Mh[(size_t)n * C_DIM + tid + 256] = (_Float16)f1;
}

// ------------------------------------------------------------------
// 3) inner = features . clusters ; softmax -> S ; log_softmax -> cluster_probs (output)
// ------------------------------------------------------------------
__global__ __launch_bounds__(256) void k_inner(const float* __restrict__ Mf,
                                               const float* __restrict__ nc,
                                               const float* __restrict__ alpha_p,
                                               float* __restrict__ Sf,
                                               _Float16* __restrict__ Sh,
                                               float* __restrict__ maxinner,
                                               float* __restrict__ out_probs) {
  __shared__ float row[C_DIM];
  __shared__ float innerL[K_CLS];
  const int n = blockIdx.x, tid = threadIdx.x;
  row[tid]       = Mf[(size_t)n * C_DIM + tid];
  row[tid + 256] = Mf[(size_t)n * C_DIM + tid + 256];
  __syncthreads();
  if (tid < K_CLS) {
    const float* cr = nc + tid * C_DIM;
    float s = 0.f;
#pragma unroll 4
    for (int c = 0; c < C_DIM; ++c) s += row[c] * cr[c];
    innerL[tid] = s;
  }
  __syncthreads();
  if (tid < 32) {
    const float alpha = alpha_p[0];
    float mxRaw = -1e30f, mxS = -1e30f;
    for (int k = 0; k < K_CLS; ++k) {
      mxRaw = fmaxf(mxRaw, innerL[k]);
      mxS   = fmaxf(mxS, alpha * innerL[k]);
    }
    float se = 0.f;
    for (int k = 0; k < K_CLS; ++k) se += expf(alpha * innerL[k] - mxS);
    const float lse = mxS + logf(se);
    const int b = n >> 12, w = (n >> 6) & 63, h = n & 63;
    if (tid < K_CLS) {
      float lp = alpha * innerL[tid] - lse;          // log_softmax
      float sm = expf(lp);                           // softmax
      Sf[(size_t)n * K_PAD + tid] = sm;
      Sh[(size_t)n * K_PAD + tid] = (_Float16)sm;
      out_probs[((size_t)(b * K_CLS + tid) * H_DIM + h) * W_DIM + w] = lp;
    } else {
      Sf[(size_t)n * K_PAD + tid] = 0.f;
      Sh[(size_t)n * K_PAD + tid] = (_Float16)0.f;
    }
    if (tid == 0) maxinner[n] = mxRaw;
  }
}

// ------------------------------------------------------------------
// 4) main pass: P = relu(M M^T), diag=0 (WMMA f16 tiles).
//    accumulates Deg (row sums), and tr(S^T P S) via G = S S^T (one extra WMMA/tile).
//    Each wave owns a 32x32 macro-tile, loops over j-chunks.
// ------------------------------------------------------------------
__global__ __launch_bounds__(256) void k_modularity(const _Float16* __restrict__ Mh,
                                                    const _Float16* __restrict__ Sh,
                                                    float* __restrict__ Deg,
                                                    float* __restrict__ modsum) {
  const int tid = threadIdx.x;
  const int lane = tid & 31, wv = tid >> 5;
  const int half = lane >> 4, nIdx = lane & 15;
  const int i0 = blockIdx.x * 32;

  // persistent S row-fragments for this wave's 32 rows (K padded to 32)
  v16h sa[2];
  sa[0] = load_fragA(Sh, K_PAD, i0,      0, lane);
  sa[1] = load_fragA(Sh, K_PAD, i0 + 16, 0, lane);

  float degp[16];
#pragma unroll
  for (int u = 0; u < 16; ++u) degp[u] = 0.f;
  float modAcc = 0.f;

  const int jStride = 32 * 8 * gridDim.y;
  for (int j0 = (blockIdx.y * 8 + wv) * 32; j0 < N_PIX; j0 += jStride) {
    v8f acc[2][2];
#pragma unroll
    for (int ti = 0; ti < 2; ++ti)
#pragma unroll
      for (int tj = 0; tj < 2; ++tj) acc[ti][tj] = (v8f){};

    for (int kc = 0; kc < C_DIM; kc += 32) {
      v16h a0 = load_fragA(Mh, C_DIM, i0,      kc, lane);
      v16h a1 = load_fragA(Mh, C_DIM, i0 + 16, kc, lane);
      v16h b0 = load_fragB(Mh, C_DIM, j0,      kc, lane);
      v16h b1 = load_fragB(Mh, C_DIM, j0 + 16, kc, lane);
      acc[0][0] = wmma_f16(a0, b0, acc[0][0]);
      acc[0][1] = wmma_f16(a0, b1, acc[0][1]);
      acc[1][0] = wmma_f16(a1, b0, acc[1][0]);
      acc[1][1] = wmma_f16(a1, b1, acc[1][1]);
    }

    // G tiles: S_i S_j^T  (16x16x32, zero C)
    v16h sb0 = load_fragB(Sh, K_PAD, j0,      0, lane);
    v16h sb1 = load_fragB(Sh, K_PAD, j0 + 16, 0, lane);
    v8f z = (v8f){};
    v8f g[2][2];
    g[0][0] = wmma_f16(sa[0], sb0, z);
    g[0][1] = wmma_f16(sa[0], sb1, z);
    g[1][0] = wmma_f16(sa[1], sb0, z);
    g[1][1] = wmma_f16(sa[1], sb1, z);

#pragma unroll
    for (int ti = 0; ti < 2; ++ti)
#pragma unroll
      for (int tj = 0; tj < 2; ++tj) {
        const int rb = i0 + 16 * ti + 8 * half;
        const int col = j0 + 16 * tj + nIdx;
#pragma unroll
        for (int v = 0; v < 8; ++v) {
          float e = acc[ti][tj][v];
          e = e > 0.f ? e : 0.f;                 // relu
          if (rb + v == col) e = 0.f;            // zero diagonal
          degp[ti * 8 + v] += e;
          modAcc += e * g[ti][tj][v];
        }
      }
  }

  // Deg: reduce across the 16 lanes of each half (cols), then atomic per row
#pragma unroll
  for (int u = 0; u < 16; ++u) {
    float s = degp[u];
    s += __shfl_xor(s, 1, 32);
    s += __shfl_xor(s, 2, 32);
    s += __shfl_xor(s, 4, 32);
    s += __shfl_xor(s, 8, 32);
    if (nIdx == 0)
      atomicAdd(&Deg[i0 + 16 * (u >> 3) + 8 * half + (u & 7)], s);
  }
  // modularity scalar: full-wave reduce
  modAcc += __shfl_xor(modAcc, 1, 32);
  modAcc += __shfl_xor(modAcc, 2, 32);
  modAcc += __shfl_xor(modAcc, 4, 32);
  modAcc += __shfl_xor(modAcc, 8, 32);
  modAcc += __shfl_xor(modAcc, 16, 32);
  if (lane == 0) atomicAdd(modsum, modAcc);
}

// ------------------------------------------------------------------
// 5) StS = S^T S (27x27)
// ------------------------------------------------------------------
__global__ __launch_bounds__(256) void k_sts(const float* __restrict__ Sf,
                                             float* __restrict__ StS) {
  __shared__ float red[256];
  const int k1 = blockIdx.x / K_CLS, k2 = blockIdx.x % K_CLS;
  float p = 0.f;
  for (int n = threadIdx.x; n < N_PIX; n += 256)
    p += Sf[(size_t)n * K_PAD + k1] * Sf[(size_t)n * K_PAD + k2];
  float s = block_reduce256(p, red);
  if (threadIdx.x == 0) StS[blockIdx.x] = s;
}

// ------------------------------------------------------------------
// 6) final: assemble losses
// ------------------------------------------------------------------
__global__ __launch_bounds__(256) void k_final(const float* __restrict__ Deg,
                                               const float* __restrict__ Sf,
                                               const float* __restrict__ maxinner,
                                               const float* __restrict__ StS,
                                               const float* __restrict__ modsum,
                                               float* __restrict__ out) {
  __shared__ float red[256];
  const int tid = threadIdx.x;

  // degsum = 2*norm ; summax
  float pd = 0.f, pm = 0.f;
  for (int n = tid; n < N_PIX; n += 256) { pd += Deg[n]; pm += maxinner[n]; }
  const float degsum = block_reduce256(pd, red);
  const float summax = block_reduce256(pm, red);

  // t = S^T Deg, accumulate ||t||^2
  float t[K_CLS];
#pragma unroll
  for (int k = 0; k < K_CLS; ++k) t[k] = 0.f;
  for (int n = tid; n < N_PIX; n += 256) {
    float d = Deg[n];
    const float* sp = Sf + (size_t)n * K_PAD;
#pragma unroll
    for (int k = 0; k < K_CLS; ++k) t[k] += sp[k] * d;
  }
  __shared__ float tn2_s;
  if (tid == 0) tn2_s = 0.f;
  __syncthreads();
  for (int k = 0; k < K_CLS; ++k) {
    float tk = block_reduce256(t[k], red);
    if (tid == 0) tn2_s += tk * tk;
  }
  __syncthreads();
  const float tn2 = tn2_s;

  // ortho loss from StS
  float pf = 0.f;
  for (int i = tid; i < K_CLS * K_CLS; i += 256) pf += StS[i] * StS[i];
  const float fro = sqrtf(block_reduce256(pf, red));
  const float invsqK = 1.0f / sqrtf((float)K_CLS);
  float po = 0.f;
  for (int i = tid; i < K_CLS * K_CLS; i += 256) {
    float v = StS[i] / fro - ((i / K_CLS == i % K_CLS) ? invsqK : 0.f);
    po += v * v;
  }
  const float ortho = sqrtf(block_reduce256(po, red));

  if (tid == 0) {
    const float twoNorm = degsum;               // 2 * norm
    const float modP = modsum[0];               // tr(S^T P S)
    const float soft_mod = -(modP - RESOLUTION * tn2 / twoNorm) / twoNorm;
    const float cluster_loss = -summax / (float)N_PIX;
    out[0] = soft_mod + cluster_loss + ortho;
    out[1] = soft_mod;
  }
}

// ------------------------------------------------------------------
extern "C" void kernel_launch(void* const* d_in, const int* in_sizes, int n_in,
                              void* d_out, int out_size, void* d_ws, size_t ws_size,
                              hipStream_t stream) {
  (void)in_sizes; (void)n_in; (void)out_size; (void)ws_size;
  const float* x        = (const float*)d_in[0];
  const float* clusters = (const float*)d_in[1];
  const float* alpha    = (const float*)d_in[2];
  float* out = (float*)d_out;

  char* ws = (char*)d_ws;
  _Float16* Mh = (_Float16*)(ws + OFF_MH);
  float*    Mf = (float*)(ws + OFF_MF);
  _Float16* Sh = (_Float16*)(ws + OFF_SH);
  float*    Sf = (float*)(ws + OFF_SF);
  float*    nc = (float*)(ws + OFF_NC);
  float*    mx = (float*)(ws + OFF_MAX);
  float*    StS = (float*)(ws + OFF_STS);
  float*    Deg = (float*)(ws + OFF_DEG);
  float*    mod = (float*)(ws + OFF_MOD);

  // zero the atomic accumulators (Deg + modsum) each call
  hipMemsetAsync(ws + OFF_DEG, 0, 32768 + 256, stream);

  k_cluster_norm<<<K_CLS, 256, 0, stream>>>(clusters, nc);
  k_feat_norm<<<N_PIX, 256, 0, stream>>>(x, Mh, Mf);
  k_inner<<<N_PIX, 256, 0, stream>>>(Mf, nc, alpha, Sf, Sh, mx, out + 2);
  k_modularity<<<dim3(N_PIX / 32, 4), 256, 0, stream>>>(Mh, Sh, Deg, mod);
  k_sts<<<K_CLS * K_CLS, 256, 0, stream>>>(Sf, StS);
  k_final<<<1, 256, 0, stream>>>(Deg, Sf, mx, StS, mod, out);
}
